// TextClassification_84026740179282
// MI455X (gfx1250) — compile-verified, matrix operand order checked by reading, not run on hardware
//
#include <hip/hip_runtime.h>
#include <math.h>

typedef __attribute__((ext_vector_type(16))) _Float16 v16h;
typedef __attribute__((ext_vector_type(8)))  _Float16 v8h;
typedef __attribute__((ext_vector_type(8)))  float    v8f;

// ---------------------------------------------------------------------------
// Fragment loaders: lane already selected its row pointer (row-major, K
// contiguous). Elements 0..7 = K[hi*8 .. hi*8+7], 8..15 = K[16+hi*8 ..].
// Matches CDNA5 16-bit A/B 16x32 VGPR striping (CDNA5 ISA 7.12.2).
// ---------------------------------------------------------------------------
__device__ __forceinline__ v16h ld_frag32(const _Float16* __restrict__ p, int hi) {
    v8h lo = *(const v8h*)(p + hi * 8);
    v8h hv = *(const v8h*)(p + 16 + hi * 8);
    v16h r;
#pragma unroll
    for (int j = 0; j < 8; ++j) { r[j] = lo[j]; r[j + 8] = hv[j]; }
    return r;
}

// Only K=0..15 valid (hd=16 attention); upper half zero.
__device__ __forceinline__ v16h ld_frag16z(const _Float16* __restrict__ p, int hi) {
    v8h lo = *(const v8h*)(p + hi * 8);
    v16h r;
#pragma unroll
    for (int j = 0; j < 8; ++j) { r[j] = lo[j]; r[j + 8] = (_Float16)0.0f; }
    return r;
}

// ---------------------------------------------------------------------------
// Generic WMMA GEMM:  C[M,N] = X[M,K](f16) @ W(K,N) + bias, W stored
// transposed as WT[N,K] f16.
//
// Block = 256 threads (8 waves). Block computes a 128x64 output tile:
// wave w owns rows [w*16, w*16+16).  The 64x32 B tile for the current K-step
// is staged in LDS via global_load_async_to_lds_b128 (ASYNCcnt), double
// buffered, and shared by all 8 waves -> 8x less L2 weight traffic.
//
// MODE 0: store f32   1: store f16   2: store f16 + ReLU
// MODE 3: store f16 transposed as [b][n][s]  (for V of attention)
// ---------------------------------------------------------------------------
template <int MODE>
__global__ __launch_bounds__(256) void gemm_wmma_kernel(
        const _Float16* __restrict__ X,
        const _Float16* __restrict__ WT,
        const float* __restrict__ bias,
        float* __restrict__ Cf,
        _Float16* __restrict__ Ch,
        int M, int N, int K, int Sdim) {
    __shared__ __align__(16) _Float16 Blds[2][64 * 32];

    const int lane = threadIdx.x & 31;
    const int wave = threadIdx.x >> 5;
    const int nt64 = N >> 6;
    const int bm   = blockIdx.x / nt64;
    const int tn   = blockIdx.x - bm * nt64;
    const int mBase = bm * 128 + wave * 16;
    const int nBase = tn << 6;
    const int mrow = lane & 15;
    const int hi   = lane >> 4;

    // async B-tile copy: 64 rows x 32 k-halves = 4KB = 256 threads x 16B
    const int n_local = threadIdx.x >> 2;        // 0..63
    const int kc      = (threadIdx.x & 3) * 8;   // half-offset 0,8,16,24
    auto issue_b = [&](int buf, int k0) {
        const _Float16* gsrc = WT + (size_t)(nBase + n_local) * K + k0 + kc;
        unsigned ldst = (unsigned)(uintptr_t)(&Blds[buf][n_local * 32 + kc]);
        asm volatile("global_load_async_to_lds_b128 %0, %1, off"
                     :: "v"(ldst), "v"((unsigned long long)(uintptr_t)gsrc)
                     : "memory");
    };
    issue_b(0, 0);

    v8f acc[4];
#pragma unroll
    for (int t = 0; t < 4; ++t)
#pragma unroll
        for (int j = 0; j < 8; ++j) acc[t][j] = 0.0f;

    const _Float16* arow = X + (size_t)(mBase + mrow) * K;
    int cur = 0;
    for (int k0 = 0; k0 < K; k0 += 32) {
        asm volatile("s_wait_asynccnt 0x0" ::: "memory");
        __syncthreads();                       // B tile [cur] visible to all waves
        if (k0 + 32 < K) issue_b(cur ^ 1, k0 + 32);

        v16h a = ld_frag32(arow + k0, hi);
        __builtin_prefetch(arow + k0 + 64, 0, 1);
#pragma unroll
        for (int t = 0; t < 4; ++t) {
            const _Float16* bp = &Blds[cur][(t * 16 + mrow) * 32];
            v8h lo = *(const v8h*)(bp + hi * 8);
            v8h hv = *(const v8h*)(bp + 16 + hi * 8);
            v16h b;
#pragma unroll
            for (int j = 0; j < 8; ++j) { b[j] = lo[j]; b[j + 8] = hv[j]; }
            acc[t] = __builtin_amdgcn_wmma_f32_16x16x32_f16(
                false, a, false, b, (short)0, acc[t], false, false);
        }
        cur ^= 1;
    }

    const int rb = mBase + (hi << 3);
#pragma unroll
    for (int t = 0; t < 4; ++t) {
        const int n  = nBase + t * 16 + mrow;
        const float bv = bias ? bias[n] : 0.0f;
#pragma unroll
        for (int j = 0; j < 8; ++j) {
            float v = acc[t][j] + bv;
            const int m = rb + j;
            if (MODE == 0) {
                Cf[(size_t)m * N + n] = v;
            } else if (MODE == 1) {
                Ch[(size_t)m * N + n] = (_Float16)v;
            } else if (MODE == 2) {
                Ch[(size_t)m * N + n] = (_Float16)fmaxf(v, 0.0f);
            } else {  // transposed store: m = b*S+s, n = d  ->  vT[b][n][s]
                const int bb = m / Sdim;
                const int s  = m - bb * Sdim;
                Ch[((size_t)bb * N + n) * Sdim + s] = (_Float16)v;
            }
        }
    }
}

// ---------------------------------------------------------------------------
// Flash attention, one wave per 16-row Q tile, key blocks of 32.
// Q,K: [b*S+s][D] f16 (head slice at head*HD).  Vt: [b][D][S] f16.
// Online softmax; P transposed C->A layout through per-wave LDS.
// ---------------------------------------------------------------------------
template <int HD>
__global__ void flash_attn_kernel(const _Float16* __restrict__ Q,
                                  const _Float16* __restrict__ Km,
                                  const _Float16* __restrict__ Vt,
                                  float* __restrict__ O,
                                  int S, int D, int H, float scale) {
    __shared__ __align__(16) _Float16 Plds[4 * 16 * 32];
    const int lane = threadIdx.x & 31;
    const int wave = threadIdx.x >> 5;
    const int qtiles = S >> 4;
    int wid = blockIdx.x * 4 + wave;
    const int qt = wid % qtiles;  wid /= qtiles;
    const int head = wid % H;
    const int b    = wid / H;
    const int mrow = lane & 15;
    const int hi   = lane >> 4;
    _Float16* pbuf = &Plds[wave * 16 * 32];

    const _Float16* qrow  = Q  + ((size_t)(b * S + qt * 16 + mrow)) * D + head * HD;
    const _Float16* kbase = Km + ((size_t)b * S) * D + head * HD;
    const _Float16* vbase = Vt + ((size_t)b * D + head * HD) * S;

    v16h aq = (HD == 32) ? ld_frag32(qrow, hi) : ld_frag16z(qrow, hi);

    float mj[8], lj[8];
    v8f acc0, acc1;
#pragma unroll
    for (int j = 0; j < 8; ++j) { mj[j] = -1e30f; lj[j] = 0.0f; acc0[j] = 0.0f; acc1[j] = 0.0f; }

    for (int kb = 0; kb < S; kb += 32) {
        const _Float16* k0p = kbase + (size_t)(kb + mrow) * D;
        const _Float16* k1p = kbase + (size_t)(kb + 16 + mrow) * D;
        v16h b0 = (HD == 32) ? ld_frag32(k0p, hi) : ld_frag16z(k0p, hi);
        v16h b1 = (HD == 32) ? ld_frag32(k1p, hi) : ld_frag16z(k1p, hi);

        v8f z;
#pragma unroll
        for (int j = 0; j < 8; ++j) z[j] = 0.0f;
        v8f s0 = __builtin_amdgcn_wmma_f32_16x16x32_f16(false, aq, false, b0, (short)0, z, false, false);
        v8f s1 = __builtin_amdgcn_wmma_f32_16x16x32_f16(false, aq, false, b1, (short)0, z, false, false);

#pragma unroll
        for (int j = 0; j < 8; ++j) {
            float t0 = s0[j] * scale;
            float t1 = s1[j] * scale;
            float mx = fmaxf(t0, t1);
#pragma unroll
            for (int off = 1; off < 16; off <<= 1) mx = fmaxf(mx, __shfl_xor(mx, off, 16));
            float mnew = fmaxf(mj[j], mx);
            float corr = __expf(mj[j] - mnew);
            float p0 = __expf(t0 - mnew);
            float p1 = __expf(t1 - mnew);
            float rs = p0 + p1;
#pragma unroll
            for (int off = 1; off < 16; off <<= 1) rs += __shfl_xor(rs, off, 16);
            mj[j] = mnew;
            lj[j] = lj[j] * corr + rs;
            acc0[j] *= corr;
            if (HD == 32) acc1[j] *= corr;
            const int row = (hi << 3) + j;
            pbuf[row * 32 + mrow]      = (_Float16)p0;
            pbuf[row * 32 + 16 + mrow] = (_Float16)p1;
        }

        // P fragment (A layout); DS ops are in-order within a wave.
        v16h ap;
        {
            v8h lo = *(const v8h*)(pbuf + mrow * 32 + hi * 8);
            v8h hv = *(const v8h*)(pbuf + mrow * 32 + 16 + hi * 8);
#pragma unroll
            for (int j = 0; j < 8; ++j) { ap[j] = lo[j]; ap[j + 8] = hv[j]; }
        }
        v16h bv0 = ld_frag32(vbase + (size_t)mrow * S + kb, hi);
        acc0 = __builtin_amdgcn_wmma_f32_16x16x32_f16(false, ap, false, bv0, (short)0, acc0, false, false);
        if (HD == 32) {
            v16h bv1 = ld_frag32(vbase + (size_t)(16 + mrow) * S + kb, hi);
            acc1 = __builtin_amdgcn_wmma_f32_16x16x32_f16(false, ap, false, bv1, (short)0, acc1, false, false);
        }
    }

    float* obase = O + ((size_t)(b * S + qt * 16)) * D + head * HD;
    const int rb = hi << 3;
#pragma unroll
    for (int j = 0; j < 8; ++j) {
        float inv = 1.0f / lj[j];
        obase[(size_t)(rb + j) * D + mrow] = acc0[j] * inv;
        if (HD == 32) obase[(size_t)(rb + j) * D + 16 + mrow] = acc1[j] * inv;
    }
}

// ---------------------------------------------------------------------------
// Elementwise / reduction helpers
// ---------------------------------------------------------------------------
__global__ void addpe_kernel(const float* __restrict__ x, float* __restrict__ r, int n) {
    int i = blockIdx.x * blockDim.x + threadIdx.x;
    if (i >= n) return;
    int d = i & 255;
    int s = (i >> 8) & 2047;
    // ang = s * 10000^(-d/256) = s * exp(-d * ln(10000)/256)
    float ang = (float)s * __expf(-(float)d * (9.210340371976184f / 256.0f));
    float pe = (d & 1) ? __cosf(ang) : __sinf(ang);
    r[i] = x[i] + pe;
}

__global__ void cvt16_kernel(const float* __restrict__ src, _Float16* __restrict__ dst, int n) {
    int i = blockIdx.x * blockDim.x + threadIdx.x;
    if (i < n) dst[i] = (_Float16)src[i];
}

// wt[n*K+k] = w[k*N+n]  (f32 [K,N] -> f16 [N,K])
__global__ void wtrans_kernel(const float* __restrict__ w, _Float16* __restrict__ wt, int K, int N) {
    int i = blockIdx.x * blockDim.x + threadIdx.x;
    if (i >= K * N) return;
    int k = i % K;
    int n = i / K;
    wt[(size_t)n * K + k] = (_Float16)w[(size_t)k * N + n];
}

// LayerNorm with fused residual: dst = LN(a + resid) * g + b ; one wave/row.
template <int DLN>
__global__ void ln_kernel(float* __restrict__ dst, const float* __restrict__ a,
                          const float* __restrict__ resid,
                          const float* __restrict__ g, const float* __restrict__ be,
                          int Mtot) {
    const int lane = threadIdx.x & 31;
    const int row  = blockIdx.x * (blockDim.x >> 5) + (threadIdx.x >> 5);
    if (row >= Mtot) return;
    const float* ap = a + (size_t)row * DLN;
    const float* rp = resid + (size_t)row * DLN;
    float xv[DLN / 32];
    float s = 0.0f;
#pragma unroll
    for (int i = 0; i < DLN / 32; ++i) {
        float v = ap[lane + (i << 5)] + rp[lane + (i << 5)];
        xv[i] = v;
        s += v;
    }
#pragma unroll
    for (int off = 1; off < 32; off <<= 1) s += __shfl_xor(s, off, 32);
    float mean = s * (1.0f / DLN);
    float vs = 0.0f;
#pragma unroll
    for (int i = 0; i < DLN / 32; ++i) { float d = xv[i] - mean; vs += d * d; }
#pragma unroll
    for (int off = 1; off < 32; off <<= 1) vs += __shfl_xor(vs, off, 32);
    float rstd = rsqrtf(vs * (1.0f / DLN) + 1e-5f);
    float* dp = dst + (size_t)row * DLN;
#pragma unroll
    for (int i = 0; i < DLN / 32; ++i) {
        int c = lane + (i << 5);
        dp[c] = (xv[i] - mean) * rstd * g[c] + be[c];
    }
}

// max over sequence: in [8,2048,64] -> out [8,64]
__global__ void maxpool_kernel(const float* __restrict__ in, float* __restrict__ out) {
    int i = blockIdx.x * blockDim.x + threadIdx.x;
    if (i >= 8 * 64) return;
    int b = i >> 6, n = i & 63;
    const float* p = in + ((size_t)b * 2048) * 64 + n;
    float m = -3.4e38f;
    for (int s = 0; s < 2048; ++s) m = fmaxf(m, p[(size_t)s * 64]);
    out[i] = m;
}

// pooled[8,64] @ w3[64,32]+b3 @ w4[32,14]+b4 -> out[8,14]
__global__ void head_kernel(const float* __restrict__ pooled,
                            const float* __restrict__ w3, const float* __restrict__ b3,
                            const float* __restrict__ w4, const float* __restrict__ b4,
                            float* __restrict__ out) {
    __shared__ float h3[8 * 32];
    int tid = threadIdx.x;
    if (tid < 8 * 32) {
        int b = tid >> 5, j = tid & 31;
        float s = b3[j];
        for (int k = 0; k < 64; ++k) s += pooled[b * 64 + k] * w3[k * 32 + j];
        h3[tid] = s;
    }
    __syncthreads();
    if (tid < 8 * 14) {
        int b = tid / 14, o = tid - b * 14;
        float s = b4[o];
        for (int j = 0; j < 32; ++j) s += h3[b * 32 + j] * w4[j * 14 + o];
        out[b * 14 + o] = s;
    }
}

// ---------------------------------------------------------------------------
// Host-side orchestration
// ---------------------------------------------------------------------------
static void launch_gemm(int mode, const _Float16* X, const _Float16* WT, const float* bias,
                        float* Cf, _Float16* Ch, int M, int N, int K, int Sdim,
                        hipStream_t st) {
    int blocks = (M / 128) * (N / 64);
    switch (mode) {
    case 0: gemm_wmma_kernel<0><<<blocks, 256, 0, st>>>(X, WT, bias, Cf, Ch, M, N, K, Sdim); break;
    case 1: gemm_wmma_kernel<1><<<blocks, 256, 0, st>>>(X, WT, bias, Cf, Ch, M, N, K, Sdim); break;
    case 2: gemm_wmma_kernel<2><<<blocks, 256, 0, st>>>(X, WT, bias, Cf, Ch, M, N, K, Sdim); break;
    default: gemm_wmma_kernel<3><<<blocks, 256, 0, st>>>(X, WT, bias, Cf, Ch, M, N, K, Sdim); break;
    }
}

struct EncW {
    const _Float16 *WqT, *WkT, *WvT, *WoT, *Wf1T, *Wf2T;
    const float *bq, *bk, *bv, *bo, *bf1, *bf2, *g1, *be1, *g2, *be2;
};

static void run_encoder(int d, int ff, int hd, const EncW& w,
                        float* r, float* t, _Float16* a16, _Float16* hid16,
                        _Float16* q16, _Float16* k16, _Float16* vT16,
                        int M, int S, hipStream_t st) {
    const int nElem = M * d;
    const int cb = (nElem + 255) / 256;
    const int H = d / hd;
    const float scale = 1.0f / sqrtf((float)hd);

    cvt16_kernel<<<cb, 256, 0, st>>>(r, a16, nElem);
    launch_gemm(1, a16, w.WqT, w.bq, nullptr, q16,  M, d, d, S, st);
    launch_gemm(1, a16, w.WkT, w.bk, nullptr, k16,  M, d, d, S, st);
    launch_gemm(3, a16, w.WvT, w.bv, nullptr, vT16, M, d, d, S, st);

    int fwaves = (M / S) * H * (S >> 4);
    int fblocks = fwaves >> 2;
    if (hd == 32)
        flash_attn_kernel<32><<<fblocks, 128, 0, st>>>(q16, k16, vT16, t, S, d, H, scale);
    else
        flash_attn_kernel<16><<<fblocks, 128, 0, st>>>(q16, k16, vT16, t, S, d, H, scale);

    cvt16_kernel<<<cb, 256, 0, st>>>(t, a16, nElem);
    launch_gemm(0, a16, w.WoT, w.bo, t, nullptr, M, d, d, S, st);

    if (d == 256) ln_kernel<256><<<M / 4, 128, 0, st>>>(r, t, r, w.g1, w.be1, M);
    else          ln_kernel<128><<<M / 4, 128, 0, st>>>(r, t, r, w.g1, w.be1, M);

    cvt16_kernel<<<cb, 256, 0, st>>>(r, a16, nElem);
    launch_gemm(2, a16, w.Wf1T, w.bf1, nullptr, hid16, M, ff, d, S, st);
    launch_gemm(0, hid16, w.Wf2T, w.bf2, t, nullptr, M, d, ff, S, st);

    if (d == 256) ln_kernel<256><<<M / 4, 128, 0, st>>>(r, t, r, w.g2, w.be2, M);
    else          ln_kernel<128><<<M / 4, 128, 0, st>>>(r, t, r, w.g2, w.be2, M);
}

extern "C" void kernel_launch(void* const* d_in, const int* in_sizes, int n_in,
                              void* d_out, int out_size, void* d_ws, size_t ws_size,
                              hipStream_t stream) {
    (void)in_sizes; (void)n_in; (void)out_size; (void)ws_size;
    constexpr int Bb = 8, S = 2048, M = Bb * S;
    auto F = [&](int i) { return (const float*)d_in[i]; };

    size_t off = 0;
    auto carve = [&](size_t bytes) -> void* {
        void* p = (char*)d_ws + off;
        off += (bytes + 255) & ~(size_t)255;
        return p;
    };
    float*    r      = (float*)carve((size_t)M * 256 * 4);
    float*    t      = (float*)carve((size_t)M * 256 * 4);
    _Float16* a16    = (_Float16*)carve((size_t)M * 256 * 2);
    _Float16* hid16  = (_Float16*)carve((size_t)M * 1024 * 2);
    _Float16* q16    = (_Float16*)carve((size_t)M * 256 * 2);
    _Float16* k16    = (_Float16*)carve((size_t)M * 256 * 2);
    _Float16* vT16   = (_Float16*)carve((size_t)M * 256 * 2);
    float*    pooled = (float*)carve(8 * 64 * 4);
    auto carveW = [&](int K, int N) { return (_Float16*)carve((size_t)K * N * 2); };
    _Float16 *WqT1 = carveW(256, 256),  *WkT1 = carveW(256, 256);
    _Float16 *WvT1 = carveW(256, 256),  *WoT1 = carveW(256, 256);
    _Float16 *Wf1T1 = carveW(256, 1024), *Wf2T1 = carveW(1024, 256);
    _Float16 *WqT2 = carveW(128, 128),  *WkT2 = carveW(128, 128);
    _Float16 *WvT2 = carveW(128, 128),  *WoT2 = carveW(128, 128);
    _Float16 *Wf1T2 = carveW(128, 512), *Wf2T2 = carveW(512, 128);
    _Float16 *Wd1T = carveW(256, 128),  *Wd2T = carveW(128, 64);

    auto wtr = [&](const float* w, _Float16* wt, int K, int N) {
        int n = K * N;
        wtrans_kernel<<<(n + 255) / 256, 256, 0, stream>>>(w, wt, K, N);
    };
    wtr(F(1),  WqT1, 256, 256);  wtr(F(3),  WkT1, 256, 256);
    wtr(F(5),  WvT1, 256, 256);  wtr(F(7),  WoT1, 256, 256);
    wtr(F(9),  Wf1T1, 256, 1024); wtr(F(11), Wf2T1, 1024, 256);
    wtr(F(17), WqT2, 128, 128);  wtr(F(19), WkT2, 128, 128);
    wtr(F(21), WvT2, 128, 128);  wtr(F(23), WoT2, 128, 128);
    wtr(F(25), Wf1T2, 128, 512); wtr(F(27), Wf2T2, 512, 128);
    wtr(F(33), Wd1T, 256, 128);  wtr(F(35), Wd2T, 128, 64);

    // x + positional encoding
    addpe_kernel<<<(M * 256 + 255) / 256, 256, 0, stream>>>(F(0), r, M * 256);

    // encoder 1 (d=256, ff=1024, hd=32)
    EncW e1 = { WqT1, WkT1, WvT1, WoT1, Wf1T1, Wf2T1,
                F(2), F(4), F(6), F(8), F(10), F(12),
                F(13), F(14), F(15), F(16) };
    run_encoder(256, 1024, 32, e1, r, t, a16, hid16, q16, k16, vT16, M, S, stream);

    // dr1: 256 -> 128 (writes new residual stream into r; reads only a16)
    cvt16_kernel<<<(M * 256 + 255) / 256, 256, 0, stream>>>(r, a16, M * 256);
    launch_gemm(0, a16, Wd1T, F(34), r, nullptr, M, 128, 256, S, stream);

    // encoder 2 (d=128, ff=512, hd=16)
    EncW e2 = { WqT2, WkT2, WvT2, WoT2, Wf1T2, Wf2T2,
                F(18), F(20), F(22), F(24), F(26), F(28),
                F(29), F(30), F(31), F(32) };
    run_encoder(128, 512, 16, e2, r, t, a16, hid16, q16, k16, vT16, M, S, stream);

    // dr2: 128 -> 64
    cvt16_kernel<<<(M * 128 + 255) / 256, 256, 0, stream>>>(r, a16, M * 128);
    launch_gemm(0, a16, Wd2T, F(36), t, nullptr, M, 64, 128, S, stream);

    // max over sequence, then 64->32->14 head
    maxpool_kernel<<<2, 256, 0, stream>>>(t, pooled);
    head_kernel<<<1, 256, 0, stream>>>(pooled, F(37), F(38), F(39), F(40), (float*)d_out);
}
